// DGLRegressor_17145509445914
// MI455X (gfx1250) — compile-verified, matrix operand order checked by reading, not run on hardware
//
#include <hip/hip_runtime.h>
#include <hip/hip_bf16.h>

typedef __attribute__((ext_vector_type(16))) _Float16 v16h;
typedef __attribute__((ext_vector_type(8)))  float    v8f;

#define HID 128

// Hardware f32 atomic add (force global_atomic_add_f32, no CAS loop).
__device__ __forceinline__ void atomAddF32(float* p, float v) {
    asm volatile("global_atomic_add_f32 %0, %1, off" : : "v"(p), "v"(v) : "memory");
}

// ---------------- generic zero ----------------
__global__ void k_zero_f32(float* __restrict__ p, long long n) {
    long long i = (long long)blockIdx.x * blockDim.x + threadIdx.x;
    long long stride = (long long)gridDim.x * blockDim.x;
    for (; i < n; i += stride) p[i] = 0.0f;
}

// ---------------- degree accumulation ----------------
__global__ void k_degrees(const int* __restrict__ src, const int* __restrict__ dst,
                          float* __restrict__ deg_out, float* __restrict__ deg_in, int E) {
    int e = blockIdx.x * blockDim.x + threadIdx.x;
    if (e >= E) return;
    atomAddF32(&deg_out[src[e]], 1.0f);
    atomAddF32(&deg_in[dst[e]], 1.0f);
}

__global__ void k_rsqrt_clip(float* __restrict__ d, int n) {
    int i = blockIdx.x * blockDim.x + threadIdx.x;
    if (i >= n) return;
    d[i] = rsqrtf(fmaxf(d[i], 1.0f));
}

// ---------------- edge scatter: agg[dst] += h[src] * inv_out[src] ----------------
// One wave32 per edge; each lane moves a float4 slice of the 128-wide row.
__global__ void k_scatter(const float* __restrict__ hin,
                          const int* __restrict__ src, const int* __restrict__ dst,
                          const float* __restrict__ inv_out,
                          float* __restrict__ agg, int E) {
    int tid = blockIdx.x * blockDim.x + threadIdx.x;
    int e = tid >> 5;
    if (e >= E) return;
    int lane = tid & 31;
    int s = src[e], d = dst[e];
    float sc = inv_out[s];
    const float4 v = *(const float4*)(hin + (long long)s * HID + lane * 4);
    float* out = agg + (long long)d * HID + lane * 4;
    atomAddF32(out + 0, v.x * sc);
    atomAddF32(out + 1, v.y * sc);
    atomAddF32(out + 2, v.z * sc);
    atomAddF32(out + 3, v.w * sc);
}

// ---------------- W -> Wt (transpose + fp16) ----------------
__global__ void k_cvt_wt(const float* __restrict__ W, _Float16* __restrict__ Wt) {
    int i = blockIdx.x * blockDim.x + threadIdx.x;   // 16384
    int n = i >> 7, k = i & 127;
    Wt[n * HID + k] = (_Float16)W[k * HID + n];
}

// ---------------- WMMA GEMM: h = relu((agg @ W) * inv_in + b) ----------------
// Block = 256 threads = 8 waves. blockIdx.x = M-tile (16 rows); wave = N-tile (16 cols).
// K = 128 -> 4x v_wmma_f32_16x16x32_f16.
// A fragment (16-bit A 16x32, ISA 7.12.2): lane L row = L%16,
//   half j -> K = (j%8) + 16*(j/8) + 8*(L/16)  => two contiguous 8-elem runs.
// B fragment mirrors A with N across lanes; Wt is [n][k] so runs are contiguous f16.
// C/D: VGPR r -> row r + 8*(L/16), col = L%16.
__global__ void k_gemm_wmma(const float* __restrict__ agg,
                            const _Float16* __restrict__ Wt,
                            const float* __restrict__ bias,
                            const float* __restrict__ inv_in,
                            float* __restrict__ hout) {
    const int mtile = blockIdx.x;
    const int wave = threadIdx.x >> 5;
    const int lane = threadIdx.x & 31;
    const int hsel = lane >> 4;          // 0 or 1
    const int l16 = lane & 15;
    const int off = hsel * 8;
    const int arow = mtile * 16 + l16;
    const int ncol = wave * 16 + l16;

    const float*    ap = agg + (long long)arow * HID;
    const _Float16* bp = Wt + (long long)ncol * HID;

    v8f c = {};
#pragma unroll
    for (int kb = 0; kb < HID; kb += 32) {
        v16h a, b;
#pragma unroll
        for (int j = 0; j < 8; ++j) {
            a[j]     = (_Float16)ap[kb + off + j];
            a[8 + j] = (_Float16)ap[kb + off + 16 + j];
            b[j]     = bp[kb + off + j];
            b[8 + j] = bp[kb + off + 16 + j];
        }
        c = __builtin_amdgcn_wmma_f32_16x16x32_f16(false, a, false, b,
                                                   (short)0, c, false, false);
    }

    const float bcol = bias[ncol];
#pragma unroll
    for (int r = 0; r < 8; ++r) {
        int orow = mtile * 16 + r + hsel * 8;
        float v = fmaxf(c[r] * inv_in[orow] + bcol, 0.0f);
        hout[(long long)orow * HID + ncol] = v;
    }
}

// ---------------- column sums for mean pooling ----------------
__global__ void k_col_reduce(const float* __restrict__ h, float* __restrict__ colsum, int nrows) {
    int t = threadIdx.x;                 // 0..127
    float acc = 0.0f;
    for (int r = blockIdx.x; r < nrows; r += gridDim.x)
        acc += h[(long long)r * HID + t];
    atomAddF32(&colsum[t], acc);
}

// ---------------- MLP head (single block, 128 threads) ----------------
__global__ void k_head(const float* __restrict__ colsum,
                       const float* __restrict__ Wl1, const float* __restrict__ bl1,
                       const float* __restrict__ Wl2, const float* __restrict__ bl2,
                       const float* __restrict__ Wo, const float* __restrict__ bo,
                       float* __restrict__ out, float inv_n) {
    __shared__ float s0[HID], s1[HID];
    int t = threadIdx.x;
    s0[t] = colsum[t] * inv_n;
    __syncthreads();
    float acc = bl1[t];
    for (int k = 0; k < HID; ++k) acc = fmaf(s0[k], Wl1[k * HID + t], acc);
    s1[t] = fmaxf(acc, 0.0f);
    __syncthreads();
    acc = bl2[t];
    for (int k = 0; k < HID; ++k) acc = fmaf(s1[k], Wl2[k * HID + t], acc);
    s0[t] = fmaxf(acc, 0.0f) * Wo[t];
    __syncthreads();
    for (int s = 64; s > 0; s >>= 1) {
        if (t < s) s0[t] += s0[t + s];
        __syncthreads();
    }
    if (t == 0) out[0] = s0[0] + bo[0];
}

extern "C" void kernel_launch(void* const* d_in, const int* in_sizes, int n_in,
                              void* d_out, int out_size, void* d_ws, size_t ws_size,
                              hipStream_t stream) {
    const float* x   = (const float*)d_in[0];
    const int*   src = (const int*)d_in[1];
    const int*   dst = (const int*)d_in[2];
    const float* W[4]  = {(const float*)d_in[3], (const float*)d_in[5],
                          (const float*)d_in[7], (const float*)d_in[9]};
    const float* b[4]  = {(const float*)d_in[4], (const float*)d_in[6],
                          (const float*)d_in[8], (const float*)d_in[10]};
    const float* Wl1 = (const float*)d_in[11]; const float* bl1 = (const float*)d_in[12];
    const float* Wl2 = (const float*)d_in[13]; const float* bl2 = (const float*)d_in[14];
    const float* Wo  = (const float*)d_in[15]; const float* bo  = (const float*)d_in[16];

    const int N = in_sizes[0] / HID;     // 100000
    const int E = in_sizes[1];           // 1600000

    // Workspace carve-up (256B aligned)
    char* base = (char*)d_ws;
    size_t off = 0;
    auto carve = [&](size_t bytes) { char* p = base + off; off = (off + bytes + 255) & ~(size_t)255; return p; };
    float*    inv_out = (float*)carve((size_t)N * 4);
    float*    inv_in  = (float*)carve((size_t)N * 4);
    float*    agg     = (float*)carve((size_t)N * HID * 4);
    float*    h       = (float*)carve((size_t)N * HID * 4);
    _Float16* Wt      = (_Float16*)carve((size_t)HID * HID * 2);
    float*    colsum  = (float*)carve((size_t)HID * 4);
    (void)ws_size;

    // Degrees -> inv_sqrt (deg buffers are contiguous: zero+rsqrt both halves at once)
    k_zero_f32<<<1024, 256, 0, stream>>>(inv_out, 2LL * N);
    k_degrees<<<(E + 255) / 256, 256, 0, stream>>>(src, dst, inv_out, inv_in, E);
    k_rsqrt_clip<<<(2 * N + 255) / 256, 256, 0, stream>>>(inv_out, 2 * N);

    // 4 GraphConv layers
    const float* hin = x;
    const long long feat = (long long)N * HID;
    for (int layer = 0; layer < 4; ++layer) {
        k_cvt_wt<<<64, 256, 0, stream>>>(W[layer], Wt);
        k_zero_f32<<<4096, 256, 0, stream>>>(agg, feat);
        k_scatter<<<(E * 32 + 255) / 256, 256, 0, stream>>>(hin, src, dst, inv_out, agg, E);
        k_gemm_wmma<<<N / 16, 256, 0, stream>>>(agg, Wt, b[layer], inv_in, h);
        hin = h;
    }

    // Mean pool + head
    k_zero_f32<<<1, 128, 0, stream>>>(colsum, HID);
    k_col_reduce<<<512, HID, 0, stream>>>(h, colsum, N);
    k_head<<<1, HID, 0, stream>>>(colsum, Wl1, bl1, Wl2, bl2, Wo, bo,
                                  (float*)d_out, 1.0f / (float)N);
}
